// FusedProjCE_2542620639593
// MI455X (gfx1250) — compile-verified, matrix operand order checked by reading, not run on hardware
//
#include <hip/hip_runtime.h>

#define SEQ   8192
#define HID   2048
#define VOCAB 50257
#define BN    256
#define BM    128
#define KC    32
#define NK    (HID / KC)                // 64 k-steps
#define NSLAB ((VOCAB + BN - 1) / BN)   // 197
#define IGNORE_INDEX (-100)
#define NEG_INF (-__builtin_inff())

typedef __attribute__((ext_vector_type(16))) __bf16 v16bf;
typedef __attribute__((ext_vector_type(8)))  float  v8f;
typedef unsigned int v4u __attribute__((ext_vector_type(4)));
typedef int          v8i __attribute__((ext_vector_type(8)));
typedef int          v4i __attribute__((ext_vector_type(4)));

// f32 -> bf16, round to nearest even
__device__ __forceinline__ unsigned short f2bf(float f) {
    unsigned int u = __float_as_uint(f);
    u += 0x7FFFu + ((u >> 16) & 1u);
    return (unsigned short)(u >> 16);
}

// online softmax partial merge
__device__ __forceinline__ void sm_combine(float& mx, float& sm, float omx, float osm) {
    float nm = fmaxf(mx, omx);
    float e1 = (mx  == NEG_INF) ? 0.0f : __expf(mx  - nm);
    float e2 = (omx == NEG_INF) ? 0.0f : __expf(omx - nm);
    sm = sm * e1 + osm * e2;
    mx = nm;
}

// ---------------------------------------------------------------------------
// Pre-pass: f32 -> bf16 with per-32 K-group permutation [0-7][16-23][8-15][24-31]
// so GEMM-side lane fragments are single contiguous 32B reads in WMMA layout.
// One thread per 32-element K-group. Bandwidth bound; conversion VALU paid once.
// ---------------------------------------------------------------------------
__global__ __launch_bounds__(256)
void convert_bf16_perm(const float* __restrict__ src,
                       unsigned short* __restrict__ dst,
                       long long nGroups) {
    long long g = (long long)blockIdx.x * 256 + threadIdx.x;
    if (g >= nGroups) return;
    const float4* s = (const float4*)(src + g * 32);
    unsigned out[16];
    #pragma unroll
    for (int j = 0; j < 4; j++) {                 // source k-chunk of 8
        float4 a = s[2 * j];
        float4 b = s[2 * j + 1];
        float fv[8] = {a.x, a.y, a.z, a.w, b.x, b.y, b.z, b.w};
        int slot = (j == 1) ? 2 : ((j == 2) ? 1 : j);
        #pragma unroll
        for (int q = 0; q < 4; q++) {
            unsigned lo = f2bf(fv[2 * q]);
            unsigned hi = f2bf(fv[2 * q + 1]);
            out[slot * 4 + q] = lo | (hi << 16);
        }
    }
    uint4* d = (uint4*)(dst + g * 32);
    d[0] = make_uint4(out[0],  out[1],  out[2],  out[3]);
    d[1] = make_uint4(out[4],  out[5],  out[6],  out[7]);
    d[2] = make_uint4(out[8],  out[9],  out[10], out[11]);
    d[3] = make_uint4(out[12], out[13], out[14], out[15]);
}

// ---------------------------------------------------------------------------
// TDM: DMA a [tileRows x 32] bf16 tile (row stride HID elements) into LDS.
// Rows >= tensorRows are OOB and return zero (used for the last vocab slab).
// ---------------------------------------------------------------------------
__device__ __forceinline__ void tdm_load_tile(const unsigned short* base, long long elemOff,
                                              unsigned ldsAddr, int tileRows, int tensorRows) {
    unsigned long long ga = (unsigned long long)(size_t)base
                          + (unsigned long long)elemOff * 2ull;
    v4u g0;
    g0[0] = 1u;                                             // count=1, user descriptor
    g0[1] = ldsAddr;                                        // LDS byte address
    g0[2] = (unsigned)(ga & 0xFFFFFFFFull);                 // global_addr[31:0]
    g0[3] = (unsigned)((ga >> 32) & 0x1FFFFFFull)           // global_addr[56:32]
          | (2u << 30);                                     // type=2 ("image")
    v8i g1;
    g1[0] = (int)(1u << 16);                                // data_size=1 (2 bytes)
    g1[1] = (int)(((unsigned)HID & 0xFFFFu) << 16);         // tensor_dim0[15:0] @63:48
    g1[2] = (int)((((unsigned)HID) >> 16)                   // tensor_dim0[31:16] @79:64
          | (((unsigned)tensorRows & 0xFFFFu) << 16));      // tensor_dim1[15:0] @95:80
    g1[3] = (int)((((unsigned)tensorRows) >> 16)            // tensor_dim1[31:16] @111:96
          | ((unsigned)KC << 16));                          // tile_dim0=32 @127:112
    g1[4] = (int)((unsigned)tileRows & 0xFFFFu);            // tile_dim1 @143:128, tile_dim2=0
    g1[5] = (int)HID;                                       // tensor_dim0_stride[31:0] @191:160
    g1[6] = 0;                                              // stride0[47:32], stride1[15:0]
    g1[7] = 0;                                              // stride1[47:16]
    v4i g2 = {0, 0, 0, 0};                                  // 2-D: dims 2/3 unused
    v4i g3 = {0, 0, 0, 0};
    v8i g4 = {0, 0, 0, 0, 0, 0, 0, 0};                      // extra group (unused, zero)
    __builtin_amdgcn_tensor_load_to_lds(g0, g1, g2, g3, g4, 0);
}

// ---------------------------------------------------------------------------
// Kernel A: one block per 256-wide vocab slab; loops 128-token tiles.
// TDM double-buffers bf16 tiles into LDS; 8 waves do 16 WMMA each per k-step;
// epilogue does in-register + cross-lane + cross-wave online softmax.
// ---------------------------------------------------------------------------
__global__ __launch_bounds__(256)
void fused_ce_wmma(const unsigned short* __restrict__ xbf,
                   const unsigned short* __restrict__ wbf,
                   const int*   __restrict__ y,
                   float* __restrict__ pMax,
                   float* __restrict__ pSum,
                   float* __restrict__ tgt) {
    __shared__ __align__(32) unsigned short ldsA[2][BM * KC];  // 2 x 8 KB
    __shared__ __align__(32) unsigned short ldsB[2][BN * KC];  // 2 x 16 KB
    __shared__ float ldsM[4 * BM];
    __shared__ float ldsS[4 * BM];
    __shared__ int   yT[BM];

    const int tid   = threadIdx.x;
    const int lane  = tid & 31;
    const int wid   = tid >> 5;      // 0..7
    const int waveM = wid >> 2;      // 0..1  -> 64 rows each
    const int waveN = wid & 3;       // 0..3  -> 64 cols each
    const int lhalf = lane >> 4;
    const int l16   = lane & 15;
    const int nBase = blockIdx.x * BN;
    const int wRows = VOCAB - nBase;               // valid vocab rows in slab

    const unsigned ldsA0 = (unsigned)(size_t)&ldsA[0][0];
    const unsigned ldsA1 = (unsigned)(size_t)&ldsA[1][0];
    const unsigned ldsB0 = (unsigned)(size_t)&ldsB[0][0];
    const unsigned ldsB1 = (unsigned)(size_t)&ldsB[1][0];

    for (int m0 = 0; m0 < SEQ; m0 += BM) {
        v8f acc[4][4] = {};

        // prologue: DMA first k-step tiles into buffer 0
        if (wid == 0) {
            tdm_load_tile(xbf, (long long)m0 * HID,    ldsA0, BM, SEQ - m0);
            tdm_load_tile(wbf, (long long)nBase * HID, ldsB0, BN, wRows);
        }

        for (int kIdx = 0; kIdx < NK; kIdx++) {
            const int cur = kIdx & 1;
            __syncthreads();   // all waves done reading buf[cur^1] (prev iter)
            if (wid == 0) {
                if (kIdx + 1 < NK) {
                    long long kOff = (long long)(kIdx + 1) * KC;
                    tdm_load_tile(xbf, (long long)m0 * HID + kOff,
                                  cur ? ldsA0 : ldsA1, BM, SEQ - m0);
                    tdm_load_tile(wbf, (long long)nBase * HID + kOff,
                                  cur ? ldsB0 : ldsB1, BN, wRows);
                    __builtin_amdgcn_s_wait_tensorcnt(2);  // cur tiles done, next in flight
                } else {
                    __builtin_amdgcn_s_wait_tensorcnt(0);
                }
            }
            __syncthreads();   // cur tiles visible to all waves

            v16bf afrag[4], bfrag[4];
            #pragma unroll
            for (int m = 0; m < 4; m++) {
                int r = waveM * 64 + m * 16 + l16;
                afrag[m] = *(const v16bf*)&ldsA[cur][r * KC + lhalf * 16];
            }
            #pragma unroll
            for (int n = 0; n < 4; n++) {
                int r = waveN * 64 + n * 16 + l16;
                bfrag[n] = *(const v16bf*)&ldsB[cur][r * KC + lhalf * 16];
            }
            #pragma unroll
            for (int m = 0; m < 4; m++)
                #pragma unroll
                for (int n = 0; n < 4; n++)
                    acc[m][n] = __builtin_amdgcn_wmma_f32_16x16x32_bf16(
                        false, afrag[m], false, bfrag[n],
                        (short)0, acc[m][n], false, false);
        }

        // ---- epilogue: per-row online softmax over the 256-col slab ----
        if (tid < BM) yT[tid] = y[m0 + tid];
        __syncthreads();

        #pragma unroll
        for (int m = 0; m < 4; m++) {
            int rowBase = waveM * 64 + m * 16 + lhalf * 8;
            #pragma unroll
            for (int v = 0; v < 8; v++) {
                int rowLocal = rowBase + v;
                int ytv = yT[rowLocal];
                float vals[4];
                #pragma unroll
                for (int n = 0; n < 4; n++) {
                    int col = nBase + waveN * 64 + n * 16 + l16;
                    float cv = acc[m][n][v];
                    if (col < VOCAB) {
                        if (col == ytv) tgt[m0 + rowLocal] = cv;
                        vals[n] = cv;
                    } else {
                        vals[n] = NEG_INF;
                    }
                }
                float mx = fmaxf(fmaxf(vals[0], vals[1]), fmaxf(vals[2], vals[3]));
                float sm = 0.0f;
                #pragma unroll
                for (int n = 0; n < 4; n++)
                    sm += (vals[n] == NEG_INF) ? 0.0f : __expf(vals[n] - mx);
                #pragma unroll
                for (int off = 1; off < 16; off <<= 1) {
                    float omx = __shfl_xor(mx, off, 32);
                    float osm = __shfl_xor(sm, off, 32);
                    sm_combine(mx, sm, omx, osm);
                }
                if (l16 == 0) {
                    ldsM[waveN * BM + rowLocal] = mx;
                    ldsS[waveN * BM + rowLocal] = sm;
                }
            }
        }
        __syncthreads();
        if (tid < BM) {
            float mx = ldsM[tid], sm = ldsS[tid];
            #pragma unroll
            for (int q = 1; q < 4; q++)
                sm_combine(mx, sm, ldsM[q * BM + tid], ldsS[q * BM + tid]);
            size_t idx = (size_t)blockIdx.x * SEQ + (size_t)(m0 + tid);
            pMax[idx] = mx;
            pSum[idx] = sm;
        }
        __syncthreads();   // protect yT/ldsM before next m-tile reuses them
    }
}

// Kernel B: merge slab partials per token -> per-token loss
__global__ void ce_token_reduce(const float* __restrict__ pMax,
                                const float* __restrict__ pSum,
                                const float* __restrict__ tgt,
                                const int*   __restrict__ y,
                                float* __restrict__ perTok) {
    int s = blockIdx.x * blockDim.x + threadIdx.x;
    if (s >= SEQ) return;
    float mx = NEG_INF, sm = 0.0f;
    for (int q = 0; q < NSLAB; q++)
        sm_combine(mx, sm, pMax[(size_t)q * SEQ + s], pSum[(size_t)q * SEQ + s]);
    float lse = mx + logf(sm);
    int yy = y[s];
    perTok[s] = (yy != IGNORE_INDEX) ? (lse - tgt[s]) : 0.0f;
}

// Kernel C: final mean over valid tokens
__global__ void ce_final(const float* __restrict__ perTok,
                         const int*   __restrict__ y,
                         float* __restrict__ out) {
    __shared__ float sS[256];
    __shared__ float sC[256];
    float s = 0.0f, c = 0.0f;
    for (int i = threadIdx.x; i < SEQ; i += 256) {
        s += perTok[i];
        c += (y[i] != IGNORE_INDEX) ? 1.0f : 0.0f;
    }
    sS[threadIdx.x] = s; sC[threadIdx.x] = c;
    __syncthreads();
    for (int off = 128; off > 0; off >>= 1) {
        if ((int)threadIdx.x < off) {
            sS[threadIdx.x] += sS[threadIdx.x + off];
            sC[threadIdx.x] += sC[threadIdx.x + off];
        }
        __syncthreads();
    }
    if (threadIdx.x == 0) out[0] = sS[0] / sC[0];
}

extern "C" void kernel_launch(void* const* d_in, const int* in_sizes, int n_in,
                              void* d_out, int out_size, void* d_ws, size_t ws_size,
                              hipStream_t stream) {
    const float* x = (const float*)d_in[0];
    const int*   y = (const int*)d_in[1];
    const float* w = (const float*)d_in[2];

    // workspace layout
    unsigned short* xbf = (unsigned short*)d_ws;                     // SEQ*HID bf16
    unsigned short* wbf = xbf + (size_t)SEQ * HID;                   // VOCAB*HID bf16
    float* pMax   = (float*)(wbf + (size_t)VOCAB * HID);             // NSLAB*SEQ
    float* pSum   = pMax + (size_t)NSLAB * SEQ;                      // NSLAB*SEQ
    float* tgt    = pSum + (size_t)NSLAB * SEQ;                      // SEQ
    float* perTok = tgt + SEQ;                                       // SEQ

    // pre-pass conversions (paid once, bandwidth bound)
    long long xGroups = (long long)SEQ * (HID / 32);
    long long wGroups = (long long)VOCAB * (HID / 32);
    convert_bf16_perm<<<(int)((xGroups + 255) / 256), 256, 0, stream>>>(x, xbf, xGroups);
    convert_bf16_perm<<<(int)((wGroups + 255) / 256), 256, 0, stream>>>(w, wbf, wGroups);

    fused_ce_wmma<<<NSLAB, 256, 0, stream>>>(xbf, wbf, y, pMax, pSum, tgt);
    ce_token_reduce<<<SEQ / 256, 256, 0, stream>>>(pMax, pSum, tgt, y, perTok);
    ce_final<<<1, 256, 0, stream>>>(perTok, y, (float*)d_out);
}
